// ProtFill_67018669687468
// MI455X (gfx1250) — matrix operand order repaired and checked by reading, not compiled, not run on hardware
//
#include <hip/hip_runtime.h>
#include <math.h>

typedef __attribute__((ext_vector_type(16))) __bf16    v16bf;
typedef __attribute__((ext_vector_type(8)))  float     v8f;
typedef __attribute__((ext_vector_type(4)))  unsigned  u32x4;

#define BB   4
#define NN   2048
#define KK   32
#define EOUT 128
#define ROWU 212   // u32 per feature row (424 bf16, 416 used + pad)

// atom order: N=0, C=1, Ca=2, O=3, Cb=4
__device__ __constant__ int PA[24] = {0,2,4,1,1,1,0,0,4,0,2,4,2,4,2,3,1,3,0,3,4,3,3,2};
__device__ __constant__ int PB[24] = {0,2,4,0,2,4,2,4,2,1,1,1,0,0,4,3,3,1,3,0,3,2,4,3};

__device__ __forceinline__ unsigned short f2bf(float f) {
  unsigned u = __float_as_uint(f);
  u += 0x7FFFu + ((u >> 16) & 1u);          // round-to-nearest-even
  return (unsigned short)(u >> 16);
}
__device__ __forceinline__ unsigned pack2(float a, float b) {
  return (unsigned)f2bf(a) | ((unsigned)f2bf(b) << 16);
}
__device__ __forceinline__ float rbf1(float d, int m) {
  float mu = 2.0f + (20.0f / 15.0f) * (float)m;   // linspace(2,22,16)
  float t  = (d - mu) * 0.8f;                     // sigma = 1.25
  return __expf(-t * t);
}

// ---------------------------------------------------------------------------
// Kernel 0a: build atoms table [B*N][5][3] = {N, C, Ca, O, Cb}
// ---------------------------------------------------------------------------
__global__ void k_prep_atoms(const float* __restrict__ X, float* __restrict__ A) {
  int t = blockIdx.x * blockDim.x + threadIdx.x;
  if (t >= BB * NN) return;
  const float* p = X + (size_t)t * 12;
  float Nx=p[0], Ny=p[1], Nz=p[2];
  float Cx=p[3], Cy=p[4], Cz=p[5];
  float Ax=p[6], Ay=p[7], Az=p[8];
  float Ox=p[9], Oy=p[10], Oz=p[11];
  float bx=Ax-Nx, by=Ay-Ny, bz=Az-Nz;
  float cx=Cx-Ax, cy=Cy-Ay, cz=Cz-Az;
  float ax=by*cz-bz*cy, ay=bz*cx-bx*cz, az=bx*cy-by*cx;
  float Bx = -0.58273431f*ax + 0.56802827f*bx - 0.54067466f*cx + Ax;
  float By = -0.58273431f*ay + 0.56802827f*by - 0.54067466f*cy + Ay;
  float Bz = -0.58273431f*az + 0.56802827f*bz - 0.54067466f*cz + Az;
  float* o = A + (size_t)t * 15;
  o[0]=Nx; o[1]=Ny; o[2]=Nz;  o[3]=Cx; o[4]=Cy; o[5]=Cz;
  o[6]=Ax; o[7]=Ay; o[8]=Az;  o[9]=Ox; o[10]=Oy; o[11]=Oz;
  o[12]=Bx; o[13]=By; o[14]=Bz;
}

// ---------------------------------------------------------------------------
// Kernel 0b: pack W_edge (416x128 f32) into per-lane WMMA-B bf16 layout.
// slot = ((ntile*13 + kk)*32 + lane)*8 + v ; lane holds col n=ntile*16+lane%16,
// K base = (lane<16 ? 0 : 8), VGPR v<4 = K {2v, 2v+1}, v>=4 = K {8+2v, 9+2v}.
// ---------------------------------------------------------------------------
__global__ void k_prep_w(const float* __restrict__ W, unsigned* __restrict__ Wpk) {
  int idx = blockIdx.x * blockDim.x + threadIdx.x;   // 26624 total
  if (idx >= 8 * 13 * 32 * 8) return;
  int v    = idx & 7;
  int lane = (idx >> 3) & 31;
  int rest = idx >> 8;
  int kk   = rest % 13;
  int nt   = rest / 13;
  int n  = nt * 16 + (lane & 15);
  int K0 = kk * 32 + ((lane & 16) ? 8 : 0) + 2 * v + (v >= 4 ? 8 : 0);
  Wpk[idx] = pack2(W[(size_t)K0 * 128 + n], W[(size_t)(K0 + 1) * 128 + n]);
}

// ---------------------------------------------------------------------------
// Kernel 1: masked distances + top-32 per row. One wave per (b,i).
// ---------------------------------------------------------------------------
__global__ void __launch_bounds__(32)
k_topk(const float* __restrict__ X, const float* __restrict__ mask,
       float* __restrict__ DnbWS, int* __restrict__ EidxWS,
       float* __restrict__ EidxOut) {
  __shared__ float Drow[NN];
  __shared__ float M2r[NN];
  int bi = blockIdx.x;                  // b*NN + i
  int b  = bi >> 11;
  int lane = threadIdx.x;

  float cix = X[((size_t)bi * 4 + 1) * 3 + 0];
  float ciy = X[((size_t)bi * 4 + 1) * 3 + 1];
  float ciz = X[((size_t)bi * 4 + 1) * 3 + 2];
  float mi  = mask[bi];

  float lmax = 0.0f;
  for (int s = 0; s < 64; ++s) {
    int j  = s * 32 + lane;
    size_t gj = (size_t)b * NN + j;
    float dx = X[(gj * 4 + 1) * 3 + 0] - cix;
    float dy = X[(gj * 4 + 1) * 3 + 1] - ciy;
    float dz = X[(gj * 4 + 1) * 3 + 2] - ciz;
    float d  = sqrtf(dx*dx + dy*dy + dz*dz + 1e-6f);
    float m2 = mi * mask[gj];
    float Dv = m2 * d;
    Drow[j] = Dv; M2r[j] = m2;
    lmax = fmaxf(lmax, Dv);
  }
  #pragma unroll
  for (int off = 16; off >= 1; off >>= 1)
    lmax = fmaxf(lmax, __shfl_xor(lmax, off, 32));
  for (int s = 0; s < 64; ++s) {
    int j = s * 32 + lane;
    Drow[j] = Drow[j] + (1.0f - M2r[j]) * lmax;   // D_adjust
  }
  // 32 rounds of global-min extraction (each lane owns j%32==lane slots)
  for (int r = 0; r < KK; ++r) {
    float lv = 3.0e38f; int lj = 0x7fffffff;
    for (int s = 0; s < 64; ++s) {
      int j = s * 32 + lane;
      float v = Drow[j];
      if (v < lv || (v == lv && j < lj)) { lv = v; lj = j; }
    }
    float v = lv; int jj = lj;
    #pragma unroll
    for (int off = 16; off >= 1; off >>= 1) {
      float v2 = __shfl_xor(v, off, 32);
      int   j2 = __shfl_xor(jj, off, 32);
      if (v2 < v || (v2 == v && j2 < jj)) { v = v2; jj = j2; }
    }
    if ((jj & 31) == lane) Drow[jj] = 3.0e38f;    // owner retires winner
    if (lane == r) {
      DnbWS[(size_t)bi * KK + r]  = v;
      EidxWS[(size_t)bi * KK + r] = jj;
      EidxOut[(size_t)bi * KK + r] = (float)jj;
    }
  }
}

// ---------------------------------------------------------------------------
// Kernel 2: fused features -> bf16 WMMA GEMM (32x416 @ 416x128) -> LayerNorm
// One workgroup (256 thr = 8 waves) per (b,i).
// ---------------------------------------------------------------------------
__global__ void __launch_bounds__(256)
k_fused(const float* __restrict__ AtomsG, const int* __restrict__ EidxWS,
        const float* __restrict__ DnbWS, const int* __restrict__ Ridx,
        const int* __restrict__ Chn, const float* __restrict__ Wpos,
        const float* __restrict__ bpos, const unsigned* __restrict__ Wpk,
        const float* __restrict__ gamma, const float* __restrict__ beta,
        float* __restrict__ Eout) {
  __shared__ unsigned F4[KK * ROWU];      // 32 x 424 bf16 feature tile
  __shared__ float Cmat[KK * EOUT];       // 32 x 128 f32 GEMM result
  __shared__ float selfA[16];
  __shared__ float nbA[KK * 15];
  __shared__ int   nbIdx[KK];
  __shared__ float nbD[KK];
  __shared__ float psum[256], psum2[256];
  __shared__ float stats[64];

  int bi = blockIdx.x;
  int b  = bi >> 11;
  int i  = bi & (NN - 1);
  int tid = threadIdx.x;

  if (tid < KK) {
    nbIdx[tid] = EidxWS[(size_t)bi * KK + tid];
    nbD[tid]   = DnbWS[(size_t)bi * KK + tid];
  } else if (tid < KK + 15) {
    selfA[tid - KK] = AtomsG[(size_t)bi * 15 + (tid - KK)];
  }
  __syncthreads();

  // neighbor atoms
  for (int t = tid; t < KK * 15; t += 256) {
    int k = t / 15, c = t - k * 15;
    nbA[t] = AtomsG[(size_t)(b * NN + nbIdx[k]) * 15 + c];
  }
  // positional embedding (cols 0..15) + rbf(D_neighbors) (cols 16..31)
  {
    int k = tid >> 3, fp = tid & 7;
    int j  = nbIdx[k];
    int ri = Ridx[b * NN + i], rj = Ridx[b * NN + j];
    int ci = Chn[b * NN + i],  cj = Chn[b * NN + j];
    int d;
    if (ci == cj) { int o = ri - rj + 32; d = o < 0 ? 0 : (o > 64 ? 64 : o); }
    else d = 65;
    float e0 = Wpos[d * 16 + 2 * fp]     + bpos[2 * fp];
    float e1 = Wpos[d * 16 + 2 * fp + 1] + bpos[2 * fp + 1];
    F4[k * ROWU + fp] = pack2(e0, e1);
    float Dk = nbD[k];
    F4[k * ROWU + 8 + fp] = pack2(rbf1(Dk, 2 * fp), rbf1(Dk, 2 * fp + 1));
  }
  __syncthreads();

  // 24 atom-pair RBFs (cols 32+16p .. 47+16p)
  for (int it = tid; it < 24 * KK; it += 256) {
    int p = it >> 5, k = it & 31;                  // p uniform within a wave
    const float* Ap = &selfA[3 * PA[p]];
    const float* Bp = &nbA[k * 15 + 3 * PB[p]];
    float dx = Ap[0] - Bp[0], dy = Ap[1] - Bp[1], dz = Ap[2] - Bp[2];
    float d = sqrtf(dx*dx + dy*dy + dz*dz + 1e-6f);
    int baseu = k * ROWU + 16 + (p << 3);
    #pragma unroll
    for (int mp = 0; mp < 8; ++mp)
      F4[baseu + mp] = pack2(rbf1(d, 2 * mp), rbf1(d, 2 * mp + 1));
  }
  __syncthreads();

  // ---- WMMA GEMM: each wave owns tiles (m, nt0) and (m, nt1) ----
  int lane = tid & 31, wv = tid >> 5;
  int m   = wv & 1;
  int nt0 = wv >> 1, nt1 = nt0 + 4;
  v8f acc0 = {0,0,0,0,0,0,0,0};
  v8f acc1 = {0,0,0,0,0,0,0,0};
  int rowA = m * 16 + (lane & 15);
  int hiA  = (lane & 16) ? 4 : 0;                  // +8 bf16 K-offset
  for (int kk2 = 0; kk2 < 13; ++kk2) {
    union { unsigned u[8]; v16bf h; } a;
    int ab = rowA * ROWU + kk2 * 16 + hiA;
    // A-layout u32 offsets per ISA: v0-3 -> K{0..7}, v4-7 -> K{16..23}
    a.u[0]=F4[ab+0];  a.u[1]=F4[ab+1];  a.u[2]=F4[ab+2];  a.u[3]=F4[ab+3];
    a.u[4]=F4[ab+8];  a.u[5]=F4[ab+9];  a.u[6]=F4[ab+10]; a.u[7]=F4[ab+11];
    union { u32x4 q[2]; v16bf h; } b0, b1;
    const u32x4* p0 = (const u32x4*)(Wpk + (((nt0 * 13 + kk2) * 32 + lane) << 3));
    const u32x4* p1 = (const u32x4*)(Wpk + (((nt1 * 13 + kk2) * 32 + lane) << 3));
    b0.q[0] = p0[0]; b0.q[1] = p0[1];
    b1.q[0] = p1[0]; b1.q[1] = p1[1];
    acc0 = __builtin_amdgcn_wmma_f32_16x16x32_bf16(false, a.h, false, b0.h,
                                                   (short)0, acc0, false, false);
    acc1 = __builtin_amdgcn_wmma_f32_16x16x32_bf16(false, a.h, false, b1.h,
                                                   (short)0, acc1, false, false);
  }
  // C/D layout: VGPR r, lanes0-15 -> M=r, lanes16-31 -> M=8+r
  int crow = m * 16 + ((lane & 16) ? 8 : 0);
  int ccol = lane & 15;
  #pragma unroll
  for (int r = 0; r < 8; ++r) {
    Cmat[(crow + r) * EOUT + nt0 * 16 + ccol] = acc0[r];
    Cmat[(crow + r) * EOUT + nt1 * 16 + ccol] = acc1[r];
  }
  __syncthreads();

  // ---- LayerNorm over 128, write out ----
  int row = tid >> 3, sub = tid & 7;
  float s1 = 0.0f, s2 = 0.0f;
  #pragma unroll
  for (int c0 = 0; c0 < 16; ++c0) {
    float v = Cmat[row * EOUT + sub * 16 + c0];
    s1 += v; s2 += v * v;
  }
  psum[tid] = s1; psum2[tid] = s2;
  __syncthreads();
  if (sub == 0) {
    float a = 0.0f, q = 0.0f;
    #pragma unroll
    for (int t = 0; t < 8; ++t) { a += psum[(row << 3) + t]; q += psum2[(row << 3) + t]; }
    float mu  = a * (1.0f / 128.0f);
    float var = q * (1.0f / 128.0f) - mu * mu;
    stats[row]      = mu;
    stats[32 + row] = rsqrtf(var + 1e-5f);
  }
  __syncthreads();
  float mu = stats[row], rs = stats[32 + row];
  size_t base = ((size_t)bi * KK + row) * EOUT + sub * 16;
  #pragma unroll
  for (int c0 = 0; c0 < 16; ++c0) {
    int c = sub * 16 + c0;
    Eout[base + c0] = (Cmat[row * EOUT + c] - mu) * rs * gamma[c] + beta[c];
  }
}

// ---------------------------------------------------------------------------
extern "C" void kernel_launch(void* const* d_in, const int* in_sizes, int n_in,
                              void* d_out, int out_size, void* d_ws, size_t ws_size,
                              hipStream_t stream) {
  const float* X     = (const float*)d_in[0];
  const float* mask  = (const float*)d_in[1];
  const int*   ridx  = (const int*)d_in[2];
  const int*   chn   = (const int*)d_in[3];
  const float* Wpos  = (const float*)d_in[4];
  const float* bpos  = (const float*)d_in[5];
  const float* Wedge = (const float*)d_in[6];
  const float* gamma = (const float*)d_in[7];
  const float* beta  = (const float*)d_in[8];

  float* E     = (float*)d_out;
  float* EidxF = E + (size_t)BB * NN * KK * EOUT;    // E_idx as floats after E

  char* ws = (char*)d_ws;
  float*    AtomsG = (float*)ws;                      // 4*2048*15 f32 = 491,520 B
  int*      EidxWS = (int*)(ws + 1966080);            // 262144 i32
  float*    DnbWS  = (float*)(ws + 3014656);          // 262144 f32
  unsigned* Wpk    = (unsigned*)(ws + 4063232);       // 26624 u32

  k_prep_atoms<<<(BB * NN + 255) / 256, 256, 0, stream>>>(X, AtomsG);
  k_prep_w<<<104, 256, 0, stream>>>(Wedge, Wpk);
  k_topk<<<BB * NN, 32, 0, stream>>>(X, mask, DnbWS, EidxWS, EidxF);
  k_fused<<<BB * NN, 256, 0, stream>>>(AtomsG, EidxWS, DnbWS, ridx, chn,
                                       Wpos, bpos, Wpk, gamma, beta, E);
}
